// BDHW_LSTM_11536282157305
// MI455X (gfx1250) — matrix operand order repaired and checked by reading, not compiled, not run on hardware
//
#include <hip/hip_runtime.h>
#include <hip/hip_bf16.h>

// ---------------------------------------------------------------------------
// BDHW 2D-BiLSTM (ReNet) for MI455X / gfx1250, wave32 + bf16 WMMA.
// B=8, D=64, H=128, W=256, NH=NO=64  (4*NH = 256 gates)
//
// gx / hout intermediates stored bf16 (halves dominant HBM traffic; hout's
// only consumer converts to bf16 for WMMA anyway). Cell state c and all four
// gate accumulators live in registers in the scan kernels; only h round-trips
// through (double-buffered) LDS as the shared WMMA A operand.
// ---------------------------------------------------------------------------

#define B_   8
#define D_   64
#define H_   128
#define W_   256
#define NH_  64
#define NG_  256        // 4*NH
#define NBH  1024       // H*B  (horizontal batch)
#define NBV  2048       // W*B  (vertical batch)
#define CH_  128        // 2*NH channels after a biLSTM

typedef __bf16 bf16_t;
typedef bf16_t v16bf __attribute__((ext_vector_type(16)));
typedef float  v8f   __attribute__((ext_vector_type(8)));

#define DEV static __device__ __forceinline__

#define WMMA_BF16(A, Bm, C) \
  __builtin_amdgcn_wmma_f32_16x16x32_bf16(false, (A), false, (Bm), (short)0, (C), false, false)

DEV float sigmoidf_(float x) { return 1.0f / (1.0f + __expf(-x)); }

// ---- WMMA fragment helpers (A/B use contiguous k = 16*hi + s, wave32) -------------------

// A: 16(M) x 32(K) bf16, row-major source, leading dim ldk, K offset k0.
DEV v16bf frag_a(const bf16_t* A, int ldk, int k0) {
  int lane = threadIdx.x & 31;
  int m  = lane & 15;
  int hi = lane >> 4;
  v16bf r;
#pragma unroll
  for (int j = 0; j < 8; ++j) {
    int k = k0 + hi * 16 + 2 * j;
    r[2 * j]     = A[m * ldk + k];
    r[2 * j + 1] = A[m * ldk + k + 1];
  }
  return r;
}

// B: 32(K) x 16(N) bf16 taken as W[N][K] row-major (B = W^T tile).
DEV v16bf frag_b_bf(const bf16_t* W, int ldk, int n0, int k0) {
  int lane = threadIdx.x & 31;
  int n  = lane & 15;
  int kh = lane >> 4;
  v16bf r;
#pragma unroll
  for (int j = 0; j < 8; ++j) {
    int k = k0 + kh * 16 + 2 * j;
    r[2 * j]     = W[(n0 + n) * ldk + k];
    r[2 * j + 1] = W[(n0 + n) * ldk + k + 1];
  }
  return r;
}

DEV v16bf frag_b_f32(const float* W, int ldk, int n0, int k0) {
  int lane = threadIdx.x & 31;
  int n  = lane & 15;
  int kh = lane >> 4;
  v16bf r;
#pragma unroll
  for (int j = 0; j < 8; ++j) {
    int k = k0 + kh * 16 + 2 * j;
    r[2 * j]     = (bf16_t)W[(n0 + n) * ldk + k];
    r[2 * j + 1] = (bf16_t)W[(n0 + n) * ldk + k + 1];
  }
  return r;
}

// C/D 16x16 f32: VGPR v, lanes 0-15: (M=v, N=lane); lanes 16-31: (M=v+8, N-16)
DEV void store_c_bf(bf16_t* dst, int ldn, v8f c) {
  int lane = threadIdx.x & 31;
  int n  = lane & 15;
  int mb = (lane >> 4) * 8;
#pragma unroll
  for (int v = 0; v < 8; ++v) dst[(mb + v) * ldn + n] = (bf16_t)c[v];
}

// ---------------------------------------------------------------------------
// Horizontal input GEMM: gx[t][n][g] = x @ Wih^T + (bih+bhh)  (K=64, bf16 out)
// x[t,n,d] gathered from img[b,d,h,t], n = h*B + b
// ---------------------------------------------------------------------------
__global__ __launch_bounds__(256) void gemm_gx_h_kernel(
    const float* __restrict__ img, const float* __restrict__ Wih,
    const float* __restrict__ bih, const float* __restrict__ bhh,
    bf16_t* __restrict__ gx) {
  __shared__ bf16_t sA[16 * D_];
  const int tid  = threadIdx.x;
  const int row0 = blockIdx.x * 16;

  for (int e = tid; e < 16 * D_; e += 256) {
    int r = e >> 6, d = e & 63;
    int row = row0 + r;
    int t = row >> 10, n = row & 1023;
    int h = n >> 3, b = n & 7;
    sA[e] = (bf16_t)img[(((size_t)(b * D_ + d)) * H_ + h) * W_ + t];
  }
  __syncthreads();

  const int wave = tid >> 5;
  const int lane = tid & 31;
  bf16_t* gout = gx + (size_t)row0 * NG_;

  v16bf a0 = frag_a(sA, D_, 0);
  v16bf a1 = frag_a(sA, D_, 32);
#pragma unroll
  for (int tile = 0; tile < 2; ++tile) {
    int n0 = (wave * 2 + tile) * 16;
    float bsum = bih[n0 + (lane & 15)] + bhh[n0 + (lane & 15)];
    v8f acc;
#pragma unroll
    for (int v = 0; v < 8; ++v) acc[v] = bsum;
    acc = WMMA_BF16(a0, frag_b_f32(Wih, D_, n0, 0),  acc);
    acc = WMMA_BF16(a1, frag_b_f32(Wih, D_, n0, 32), acc);
    store_c_bf(gout + n0, NG_, acc);
  }
}

// ---------------------------------------------------------------------------
// Vertical input GEMM: vin[y,n2,c] = hout[x][y*B+b][c], n2 = x*B+b  (K=128)
// ---------------------------------------------------------------------------
__global__ __launch_bounds__(256) void gemm_gx_v_kernel(
    const bf16_t* __restrict__ hout, const float* __restrict__ Wih,
    const float* __restrict__ bih, const float* __restrict__ bhh,
    bf16_t* __restrict__ gx) {
  __shared__ bf16_t sA[16 * CH_];
  const int tid  = threadIdx.x;
  const int row0 = blockIdx.x * 16;

  for (int e = tid; e < 16 * CH_; e += 256) {
    int r = e >> 7, c = e & 127;
    int row = row0 + r;
    int y = row >> 11, n2 = row & 2047;
    int x = n2 >> 3, b = n2 & 7;
    sA[e] = hout[((size_t)x * NBH + (y * B_ + b)) * CH_ + c];
  }
  __syncthreads();

  const int wave = tid >> 5;
  const int lane = tid & 31;
  bf16_t* gout = gx + (size_t)row0 * NG_;

  v16bf a0 = frag_a(sA, CH_, 0);
  v16bf a1 = frag_a(sA, CH_, 32);
  v16bf a2 = frag_a(sA, CH_, 64);
  v16bf a3 = frag_a(sA, CH_, 96);
#pragma unroll
  for (int tile = 0; tile < 2; ++tile) {
    int n0 = (wave * 2 + tile) * 16;
    float bsum = bih[n0 + (lane & 15)] + bhh[n0 + (lane & 15)];
    v8f acc;
#pragma unroll
    for (int v = 0; v < 8; ++v) acc[v] = bsum;
    acc = WMMA_BF16(a0, frag_b_f32(Wih, CH_, n0, 0),  acc);
    acc = WMMA_BF16(a1, frag_b_f32(Wih, CH_, n0, 32), acc);
    acc = WMMA_BF16(a2, frag_b_f32(Wih, CH_, n0, 64), acc);
    acc = WMMA_BF16(a3, frag_b_f32(Wih, CH_, n0, 96), acc);
    store_c_bf(gout + n0, NG_, acc);
  }
}

// ---------------------------------------------------------------------------
// LSTM scan, 128 threads = 4 waves per WG, one WG owns 16 batch rows for all
// T steps. Wave w owns u in [16w,16w+16): it computes all four gate tiles
// (n0 = g*64 + 16w), so i/f/g/o land in this wave's four accumulators and the
// cell state c stays in registers. Only h goes through double-buffered LDS.
// ---------------------------------------------------------------------------
__global__ __launch_bounds__(128) void scan_h_kernel(
    const bf16_t* __restrict__ gx, const float* __restrict__ Whh,
    bf16_t* __restrict__ hout, int dirOff, int rev) {
  __shared__ bf16_t sW[NG_ * NH_];       // 32 KB: Whh bf16, N-major rows
  __shared__ bf16_t sH[2][16 * NH_];     // double-buffered h state

  const int tid  = threadIdx.x;
  const int row0 = blockIdx.x * 16;
  for (int e = tid; e < NG_ * NH_; e += 128) sW[e] = (bf16_t)Whh[e];
  for (int e = tid; e < 16 * NH_; e += 128) sH[0][e] = (bf16_t)0.0f;
  __syncthreads();

  const int wave = tid >> 5;
  const int lane = tid & 31;
  const int n  = lane & 15;
  const int mb = (lane >> 4) * 8;
  const int u  = wave * 16 + n;          // hidden column owned by this lane

  float creg[8];
#pragma unroll
  for (int v = 0; v < 8; ++v) creg[v] = 0.0f;

  for (int step = 0; step < W_; ++step) {
    const int t   = rev ? (W_ - 1 - step) : step;
    const int cur = step & 1, nxt = cur ^ 1;
    const bf16_t* gin = gx + ((size_t)t * NBH + row0) * NG_;
    if (step + 1 < W_) {
      const int tn = rev ? (t - 1) : (t + 1);
      __builtin_prefetch(gx + ((size_t)tn * NBH + row0) * NG_, 0, 1);
    }
    v16bf a0 = frag_a(sH[cur], NH_, 0);
    v16bf a1 = frag_a(sH[cur], NH_, 32);

    v8f acc[4];
#pragma unroll
    for (int g = 0; g < 4; ++g) {
      const int n0 = g * NH_ + wave * 16;
#pragma unroll
      for (int v = 0; v < 8; ++v)
        acc[g][v] = (float)gin[(mb + v) * NG_ + n0 + n];
      acc[g] = WMMA_BF16(a0, frag_b_bf(sW, NH_, n0, 0),  acc[g]);
      acc[g] = WMMA_BF16(a1, frag_b_bf(sW, NH_, n0, 32), acc[g]);
    }
#pragma unroll
    for (int v = 0; v < 8; ++v) {
      const int m = mb + v;
      float c = sigmoidf_(acc[1][v]) * creg[v] +
                sigmoidf_(acc[0][v]) * tanhf(acc[2][v]);
      float h = sigmoidf_(acc[3][v]) * tanhf(c);
      creg[v] = c;
      bf16_t hb = (bf16_t)h;
      sH[nxt][m * NH_ + u] = hb;
      hout[((size_t)t * NBH + row0 + m) * CH_ + dirOff + u] = hb;
    }
    __syncthreads();
  }
}

// ---------------------------------------------------------------------------
// Vertical scan over y; scatters h (f32) straight into out[b][c][y][x].
// ---------------------------------------------------------------------------
__global__ __launch_bounds__(128) void scan_v_kernel(
    const bf16_t* __restrict__ gx, const float* __restrict__ Whh,
    float* __restrict__ out, int dirOff, int rev) {
  __shared__ bf16_t sW[NG_ * NH_];
  __shared__ bf16_t sH[2][16 * NH_];

  const int tid  = threadIdx.x;
  const int row0 = blockIdx.x * 16;
  for (int e = tid; e < NG_ * NH_; e += 128) sW[e] = (bf16_t)Whh[e];
  for (int e = tid; e < 16 * NH_; e += 128) sH[0][e] = (bf16_t)0.0f;
  __syncthreads();

  const int wave = tid >> 5;
  const int lane = tid & 31;
  const int n  = lane & 15;
  const int mb = (lane >> 4) * 8;
  const int u  = wave * 16 + n;

  float creg[8];
#pragma unroll
  for (int v = 0; v < 8; ++v) creg[v] = 0.0f;

  for (int step = 0; step < H_; ++step) {
    const int y   = rev ? (H_ - 1 - step) : step;
    const int cur = step & 1, nxt = cur ^ 1;
    const bf16_t* gin = gx + ((size_t)y * NBV + row0) * NG_;
    if (step + 1 < H_) {
      const int yn = rev ? (y - 1) : (y + 1);
      __builtin_prefetch(gx + ((size_t)yn * NBV + row0) * NG_, 0, 1);
    }
    v16bf a0 = frag_a(sH[cur], NH_, 0);
    v16bf a1 = frag_a(sH[cur], NH_, 32);

    v8f acc[4];
#pragma unroll
    for (int g = 0; g < 4; ++g) {
      const int n0 = g * NH_ + wave * 16;
#pragma unroll
      for (int v = 0; v < 8; ++v)
        acc[g][v] = (float)gin[(mb + v) * NG_ + n0 + n];
      acc[g] = WMMA_BF16(a0, frag_b_bf(sW, NH_, n0, 0),  acc[g]);
      acc[g] = WMMA_BF16(a1, frag_b_bf(sW, NH_, n0, 32), acc[g]);
    }
#pragma unroll
    for (int v = 0; v < 8; ++v) {
      const int m = mb + v;
      float c = sigmoidf_(acc[1][v]) * creg[v] +
                sigmoidf_(acc[0][v]) * tanhf(acc[2][v]);
      float h = sigmoidf_(acc[3][v]) * tanhf(c);
      creg[v] = c;
      sH[nxt][m * NH_ + u] = (bf16_t)h;
      const int n2 = row0 + m;
      const int x = n2 >> 3, b = n2 & 7;
      out[(((size_t)b * CH_ + dirOff + u) * H_ + y) * W_ + x] = h;
    }
    __syncthreads();
  }
}

// ---------------------------------------------------------------------------
extern "C" void kernel_launch(void* const* d_in, const int* in_sizes, int n_in,
                              void* d_out, int out_size, void* d_ws, size_t ws_size,
                              hipStream_t stream) {
  const float* img     = (const float*)d_in[0];
  const float* hWih_f  = (const float*)d_in[1];
  const float* hWhh_f  = (const float*)d_in[2];
  const float* hbih_f  = (const float*)d_in[3];
  const float* hbhh_f  = (const float*)d_in[4];
  const float* hWih_b  = (const float*)d_in[5];
  const float* hWhh_b  = (const float*)d_in[6];
  const float* hbih_b  = (const float*)d_in[7];
  const float* hbhh_b  = (const float*)d_in[8];
  const float* vWih_f  = (const float*)d_in[9];
  const float* vWhh_f  = (const float*)d_in[10];
  const float* vbih_f  = (const float*)d_in[11];
  const float* vbhh_f  = (const float*)d_in[12];
  const float* vWih_b  = (const float*)d_in[13];
  const float* vWhh_b  = (const float*)d_in[14];
  const float* vbih_b  = (const float*)d_in[15];
  const float* vbhh_b  = (const float*)d_in[16];
  float* out = (float*)d_out;

  // workspace: gx bf16 (67,108,864 elems = 128 MiB, shared by all 4 dirs)
  //            hout bf16 (33,554,432 elems = 64 MiB)
  bf16_t* gx   = (bf16_t*)d_ws;
  bf16_t* hout = gx + (size_t)67108864;

  const dim3 gblk(256), sblk(128);
  const int gemm_blocks = (W_ * NBH) / 16;   // 16384 (same count for vertical)

  // --- horizontal biLSTM over W ---
  gemm_gx_h_kernel<<<gemm_blocks, gblk, 0, stream>>>(img, hWih_f, hbih_f, hbhh_f, gx);
  scan_h_kernel<<<NBH / 16, sblk, 0, stream>>>(gx, hWhh_f, hout, 0, 0);
  gemm_gx_h_kernel<<<gemm_blocks, gblk, 0, stream>>>(img, hWih_b, hbih_b, hbhh_b, gx);
  scan_h_kernel<<<NBH / 16, sblk, 0, stream>>>(gx, hWhh_b, hout, NH_, 1);

  // --- vertical biLSTM over H (writes straight into d_out) ---
  gemm_gx_v_kernel<<<gemm_blocks, gblk, 0, stream>>>(hout, vWih_f, vbih_f, vbhh_f, gx);
  scan_v_kernel<<<NBV / 16, sblk, 0, stream>>>(gx, vWhh_f, out, 0, 0);
  gemm_gx_v_kernel<<<gemm_blocks, gblk, 0, stream>>>(hout, vWih_b, vbih_b, vbhh_b, gx);
  scan_v_kernel<<<NBV / 16, sblk, 0, stream>>>(gx, vWhh_b, out, NH_, 1);
}
